// TPMoEExperts_15427522527439
// MI455X (gfx1250) — compile-verified, matrix operand order checked by reading, not run on hardware
//
#include <hip/hip_runtime.h>
#include <stdint.h>

#define T_TOK 8192
#define H_DIM 1024
#define I_DIM 2816
#define N_EXP 8
#define TILE_R 128
#define TILE_C 64
#define TILES_R (T_TOK / TILE_R)   // 64
#define KSTEP 32
#define LDSTR 40                    // ushort stride per LDS row (pad vs 32 -> conflict-free b128 reads)

typedef __attribute__((ext_vector_type(16))) __bf16 bf16x16;
typedef __attribute__((ext_vector_type(8)))  float  f32x8;
typedef __attribute__((ext_vector_type(4)))  int    v4i;
typedef __attribute__((address_space(1)))    v4i    v4i_glob;
typedef __attribute__((address_space(3)))    v4i    v4i_lds;

union Frag16 {
  bf16x16 v;
  uint4   q[2];
};

#if defined(__has_builtin)
#if __has_builtin(__builtin_amdgcn_global_load_async_to_lds_b128)
#define HAVE_ASYNC_LDS 1
#endif
#endif
#ifndef HAVE_ASYNC_LDS
#define HAVE_ASYNC_LDS 0
#endif

// round-half-up f32 -> bf16 (cheap: one add, then keep the high 16 bits)
__device__ __forceinline__ unsigned short f2bf(float f) {
  unsigned u = __builtin_bit_cast(unsigned, f) + 0x8000u;
  return (unsigned short)(u >> 16);
}

// pack two f32 -> one dword of bf16: 2x v_add + 1x v_perm_b32
__device__ __forceinline__ unsigned pk_bf2(float a, float b) {
  unsigned ua = __builtin_bit_cast(unsigned, a) + 0x8000u;
  unsigned ub = __builtin_bit_cast(unsigned, b) + 0x8000u;
  // dst = { hi16(ub), hi16(ua) }  (bytes: ua2,ua3,ub2,ub3)
  return __builtin_amdgcn_perm(ub, ua, 0x07060302u);
}

__device__ __forceinline__ float fast_rcp(float x) {
#if defined(__has_builtin) && __has_builtin(__builtin_amdgcn_rcpf)
  return __builtin_amdgcn_rcpf(x);
#else
  return 1.0f / x;
#endif
}

__device__ __forceinline__ void wait_async0() {
#if defined(__has_builtin) && __has_builtin(__builtin_amdgcn_s_wait_asynccnt)
  __builtin_amdgcn_s_wait_asynccnt(0);
#else
  asm volatile("s_wait_asynccnt 0x0" ::: "memory");
#endif
}

// 16-byte memory -> LDS transfer: CDNA5 async DMA when available
__device__ __forceinline__ void async16(const unsigned short* g, unsigned short* l) {
#if HAVE_ASYNC_LDS
  __builtin_amdgcn_global_load_async_to_lds_b128((v4i_glob*)(void*)g, (v4i_lds*)(void*)l, 0, 0);
#else
  *(uint4*)l = *(const uint4*)g;
#endif
}

#define WMMA_BF16(A, B, C) \
  __builtin_amdgcn_wmma_f32_16x16x32_bf16(false, (A), false, (B), (short)0, (C), false, false)

// ---------------- routing setup ----------------
// ws int layout: [0..7]=counts  [8..15]=cursor  [16..24]=offsets  [32..32+T)=perm
// then (byte offset 65536): hbuf, xb, wgb, wub, wdb (all bf16)

__global__ void moe_zero(int* ws_i) {
  if (threadIdx.x < 16) ws_i[threadIdx.x] = 0;
}

__global__ void moe_count(const int* __restrict__ idx, int* __restrict__ counts) {
  int t = blockIdx.x * 256 + threadIdx.x;
  if (t < T_TOK) atomicAdd(&counts[idx[t]], 1);
}

__global__ void moe_scan(const int* __restrict__ counts, int* __restrict__ offsets) {
  if (threadIdx.x == 0) {
    int s = 0;
    for (int e = 0; e < N_EXP; ++e) { offsets[e] = s; s += counts[e]; }
    offsets[N_EXP] = s;
  }
}

__global__ void moe_scatter(const int* __restrict__ idx, int* __restrict__ cursor,
                            const int* __restrict__ offsets, int* __restrict__ perm) {
  int t = blockIdx.x * 256 + threadIdx.x;
  if (t < T_TOK) {
    int e = idx[t];
    int p = atomicAdd(&cursor[e], 1);
    perm[offsets[e] + p] = t;
  }
}

// ---------------- f32 -> bf16 bulk conversion (8 elems/thread) ----------------

__global__ __launch_bounds__(256)
void conv_bf16(const float* __restrict__ src, unsigned short* __restrict__ dst) {
  size_t i = ((size_t)blockIdx.x * 256 + threadIdx.x) * 8;
  float4 a = *(const float4*)(src + i);
  float4 b = *(const float4*)(src + i + 4);
  uint4 o;
  o.x = pk_bf2(a.x, a.y);
  o.y = pk_bf2(a.z, a.w);
  o.z = pk_bf2(b.x, b.y);
  o.w = pk_bf2(b.z, b.w);
  *(uint4*)(dst + i) = o;
}

// ---------------- gate/up: h = silu(x Wg^T) * (x Wu^T), bf16 out ----------------

__global__ __launch_bounds__(256, 2)
void moe_gateup(const unsigned short* __restrict__ xb,
                const unsigned short* __restrict__ wgb,
                const unsigned short* __restrict__ wub,
                const int*   __restrict__ counts,
                const int*   __restrict__ offsets,
                const int*   __restrict__ perm,
                unsigned short* __restrict__ hbuf)
{
  const int e   = blockIdx.x / TILES_R;
  const int rt  = blockIdx.x % TILES_R;
  const int cnt = counts[e];
  if (rt * TILE_R >= cnt) return;
  const int base = offsets[e];
  const int ct   = blockIdx.y;              // intermediate-column tile (64 wide)

  __shared__ int sPerm[TILE_R];
  __shared__ __align__(16) unsigned short sX [2][TILE_R * LDSTR];
  __shared__ __align__(16) unsigned short sWg[2][TILE_C * LDSTR];
  __shared__ __align__(16) unsigned short sWu[2][TILE_C * LDSTR];

  const int tid = threadIdx.x;
  if (tid < TILE_R) {
    int r = rt * TILE_R + tid;
    sPerm[tid] = perm[base + (r < cnt ? r : cnt - 1)];
  }
  __syncthreads();

  const int lane  = tid & 31;
  const int wave  = tid >> 5;
  const int wr    = wave >> 1;              // 0..3  -> 32-row strip
  const int wc    = wave & 1;               // 0..1  -> 32-col strip
  const int lhalf = lane >> 4;              // 0/1
  const int lrow  = lane & 15;
  const int chunk = lhalf * 8;

  // stage coords: one 16B (8 bf16) chunk per op; row = tid>>2, chunk col = (tid&3)*8
  const int xr = tid >> 2;                  // 0..63
  const int xc = (tid & 3) * 8;             // 0/8/16/24
  int tokr[2];
  #pragma unroll
  for (int j = 0; j < 2; ++j) tokr[j] = sPerm[xr + 64 * j];

  const unsigned short* wg_base = wgb + (size_t)e * I_DIM * H_DIM + (size_t)(ct * TILE_C) * H_DIM;
  const unsigned short* wu_base = wub + (size_t)e * I_DIM * H_DIM + (size_t)(ct * TILE_C) * H_DIM;

  f32x8 cg00 = {}, cg01 = {}, cg10 = {}, cg11 = {};
  f32x8 cu00 = {}, cu01 = {}, cu10 = {}, cu11 = {};

  auto stage = [&](int k, int b) {
    #pragma unroll
    for (int j = 0; j < 2; ++j)                             // X: 128 rows, 2 ops/thread
      async16(xb + (size_t)tokr[j] * H_DIM + k + xc,
              &sX[b][(xr + 64 * j) * LDSTR + xc]);
    async16(wg_base + (size_t)xr * H_DIM + k + xc,          // Wg: 64 rows, 1 op/thread
            &sWg[b][xr * LDSTR + xc]);
    async16(wu_base + (size_t)xr * H_DIM + k + xc,          // Wu: 64 rows, 1 op/thread
            &sWu[b][xr * LDSTR + xc]);
  };
  auto compute = [&](int b) {
    const unsigned short* pX = sX[b];
    const unsigned short* pG = sWg[b];
    const unsigned short* pU = sWu[b];
    Frag16 a0, a1, bg0, bg1, bu0, bu1;
    const int ar0 = wr * 32 + lrow;
    a0.q[0] = *(const uint4*)&pX[ar0 * LDSTR + chunk];
    a0.q[1] = *(const uint4*)&pX[ar0 * LDSTR + chunk + 16];
    const int ar1 = ar0 + 16;
    a1.q[0] = *(const uint4*)&pX[ar1 * LDSTR + chunk];
    a1.q[1] = *(const uint4*)&pX[ar1 * LDSTR + chunk + 16];
    const int bc0 = wc * 32 + lrow;
    bg0.q[0] = *(const uint4*)&pG[bc0 * LDSTR + chunk];
    bg0.q[1] = *(const uint4*)&pG[bc0 * LDSTR + chunk + 16];
    bu0.q[0] = *(const uint4*)&pU[bc0 * LDSTR + chunk];
    bu0.q[1] = *(const uint4*)&pU[bc0 * LDSTR + chunk + 16];
    const int bc1 = bc0 + 16;
    bg1.q[0] = *(const uint4*)&pG[bc1 * LDSTR + chunk];
    bg1.q[1] = *(const uint4*)&pG[bc1 * LDSTR + chunk + 16];
    bu1.q[0] = *(const uint4*)&pU[bc1 * LDSTR + chunk];
    bu1.q[1] = *(const uint4*)&pU[bc1 * LDSTR + chunk + 16];

    cg00 = WMMA_BF16(a0.v, bg0.v, cg00);
    cg01 = WMMA_BF16(a0.v, bg1.v, cg01);
    cg10 = WMMA_BF16(a1.v, bg0.v, cg10);
    cg11 = WMMA_BF16(a1.v, bg1.v, cg11);
    cu00 = WMMA_BF16(a0.v, bu0.v, cu00);
    cu01 = WMMA_BF16(a0.v, bu1.v, cu01);
    cu10 = WMMA_BF16(a1.v, bu0.v, cu10);
    cu11 = WMMA_BF16(a1.v, bu1.v, cu11);
  };

  // async pipeline: DMA of tile k+1 overlaps WMMA on tile k
  const int NK = H_DIM / KSTEP;
  stage(0, 0);
  wait_async0();
  __syncthreads();
  for (int kk = 0; kk < NK; ++kk) {
    const int  cur  = kk & 1;
    const bool more = (kk + 1 < NK);
    if (more) stage((kk + 1) * KSTEP, cur ^ 1);
    compute(cur);
    if (more) wait_async0();
    __syncthreads();
  }

  // epilogue: SiLU(gate)*up -> bf16 h-buffer (count-masked rows)
  const int gcol_base = ct * TILE_C + wc * 32;
  auto emit = [&](const f32x8& g, const f32x8& u, int i, int j) {
    #pragma unroll
    for (int v2 = 0; v2 < 8; ++v2) {
      const int rl = wr * 32 + i * 16 + lhalf * 8 + v2;
      const int rglob = rt * TILE_R + rl;
      if (rglob < cnt) {
        float gv = g[v2];
        float s  = gv * fast_rcp(1.0f + __expf(-gv));
        hbuf[(size_t)(base + rglob) * I_DIM + gcol_base + j * 16 + lrow] = f2bf(s * u[v2]);
      }
    }
  };
  emit(cg00, cu00, 0, 0); emit(cg01, cu01, 0, 1);
  emit(cg10, cu10, 1, 0); emit(cg11, cu11, 1, 1);
}

// ---------------- down: out[perm[r]] = h[r] * Wd^T ----------------

__global__ __launch_bounds__(256, 2)
void moe_down(const unsigned short* __restrict__ wdb,
              const int*   __restrict__ counts,
              const int*   __restrict__ offsets,
              const int*   __restrict__ perm,
              const unsigned short* __restrict__ hbuf,
              float* __restrict__ out)
{
  const int e   = blockIdx.x / TILES_R;
  const int rt  = blockIdx.x % TILES_R;
  const int cnt = counts[e];
  if (rt * TILE_R >= cnt) return;
  const int base = offsets[e];
  const int ct   = blockIdx.y;              // H-column tile (64 wide)

  __shared__ __align__(16) unsigned short sH [2][TILE_R * LDSTR];
  __shared__ __align__(16) unsigned short sWd[2][TILE_C * LDSTR];

  const int tid   = threadIdx.x;
  const int lane  = tid & 31;
  const int wave  = tid >> 5;
  const int wr    = wave >> 1;
  const int wc    = wave & 1;
  const int lhalf = lane >> 4;
  const int lrow  = lane & 15;
  const int chunk = lhalf * 8;

  f32x8 c00 = {}, c01 = {}, c10 = {}, c11 = {};

  const unsigned short* wd_base = wdb + (size_t)e * H_DIM * I_DIM + (size_t)(ct * TILE_C) * I_DIM;

  // stage coords
  const int hr = tid >> 2;                  // 0..63
  const int hc = (tid & 3) * 8;             // 0/8/16/24

  int srow[2];
  #pragma unroll
  for (int j = 0; j < 2; ++j) {
    int rg_ = rt * TILE_R + hr + 64 * j;
    srow[j] = base + (rg_ < cnt ? rg_ : cnt - 1);
  }

  auto stage = [&](int k, int b) {
    #pragma unroll
    for (int j = 0; j < 2; ++j)                             // h: 128 rows, 2 ops/thread
      async16(hbuf + (size_t)srow[j] * I_DIM + k + hc,
              &sH[b][(hr + 64 * j) * LDSTR + hc]);
    async16(wd_base + (size_t)hr * I_DIM + k + hc,          // Wd: 64 rows, 1 op/thread
            &sWd[b][hr * LDSTR + hc]);
  };
  auto compute = [&](int b) {
    const unsigned short* pH = sH[b];
    const unsigned short* pW = sWd[b];
    Frag16 a0, a1, b0, b1;
    const int ar0 = wr * 32 + lrow;
    a0.q[0] = *(const uint4*)&pH[ar0 * LDSTR + chunk];
    a0.q[1] = *(const uint4*)&pH[ar0 * LDSTR + chunk + 16];
    const int ar1 = ar0 + 16;
    a1.q[0] = *(const uint4*)&pH[ar1 * LDSTR + chunk];
    a1.q[1] = *(const uint4*)&pH[ar1 * LDSTR + chunk + 16];
    const int bc0 = wc * 32 + lrow;
    b0.q[0] = *(const uint4*)&pW[bc0 * LDSTR + chunk];
    b0.q[1] = *(const uint4*)&pW[bc0 * LDSTR + chunk + 16];
    const int bc1 = bc0 + 16;
    b1.q[0] = *(const uint4*)&pW[bc1 * LDSTR + chunk];
    b1.q[1] = *(const uint4*)&pW[bc1 * LDSTR + chunk + 16];

    c00 = WMMA_BF16(a0.v, b0.v, c00);
    c01 = WMMA_BF16(a0.v, b1.v, c01);
    c10 = WMMA_BF16(a1.v, b0.v, c10);
    c11 = WMMA_BF16(a1.v, b1.v, c11);
  };

  const int NK = I_DIM / KSTEP;              // 88
  stage(0, 0);
  wait_async0();
  __syncthreads();
  for (int kk = 0; kk < NK; ++kk) {
    const int  cur  = kk & 1;
    const bool more = (kk + 1 < NK);
    if (more) stage((kk + 1) * KSTEP, cur ^ 1);
    compute(cur);
    if (more) wait_async0();
    __syncthreads();
  }

  // scatter f32 results to routed tokens
  const int gcol_base = ct * TILE_C + wc * 32;
  auto emit = [&](const f32x8& c, int i, int j) {
    #pragma unroll
    for (int v2 = 0; v2 < 8; ++v2) {
      const int rl = wr * 32 + i * 16 + lhalf * 8 + v2;
      const int rglob = rt * TILE_R + rl;
      if (rglob < cnt) {
        int tokid = perm[base + rglob];
        out[(size_t)tokid * H_DIM + gcol_base + j * 16 + lrow] = c[v2];
      }
    }
  };
  emit(c00, 0, 0); emit(c01, 0, 1); emit(c10, 1, 0); emit(c11, 1, 1);
}

// ---------------- host launch ----------------

extern "C" void kernel_launch(void* const* d_in, const int* in_sizes, int n_in,
                              void* d_out, int out_size, void* d_ws, size_t ws_size,
                              hipStream_t stream) {
  const float* x   = (const float*)d_in[0];
  const int*   idx = (const int*)d_in[1];
  const float* Wg  = (const float*)d_in[2];
  const float* Wu  = (const float*)d_in[3];
  const float* Wd  = (const float*)d_in[4];
  float* out = (float*)d_out;

  int* ws_i    = (int*)d_ws;
  int* counts  = ws_i;          // 8
  int* cursor  = ws_i + 8;      // 8
  int* offsets = ws_i + 16;     // 9
  int* perm    = ws_i + 32;     // T

  const size_t N_W = (size_t)N_EXP * I_DIM * H_DIM;   // 23,068,672 per weight tensor
  const size_t N_X = (size_t)T_TOK * H_DIM;           // 8,388,608
  char* p = (char*)d_ws + 65536;
  unsigned short* hbuf = (unsigned short*)p;  p += (size_t)T_TOK * I_DIM * 2;
  unsigned short* xb   = (unsigned short*)p;  p += N_X * 2;
  unsigned short* wgb  = (unsigned short*)p;  p += N_W * 2;
  unsigned short* wub  = (unsigned short*)p;  p += N_W * 2;
  unsigned short* wdb  = (unsigned short*)p;

  moe_zero<<<1, 32, 0, stream>>>(ws_i);
  moe_count<<<T_TOK / 256, 256, 0, stream>>>(idx, counts);
  moe_scan<<<1, 32, 0, stream>>>(counts, offsets);
  moe_scatter<<<T_TOK / 256, 256, 0, stream>>>(idx, cursor, offsets, perm);

  conv_bf16<<<(unsigned)(N_X / 2048), 256, 0, stream>>>(x,  xb);
  conv_bf16<<<(unsigned)(N_W / 2048), 256, 0, stream>>>(Wg, wgb);
  conv_bf16<<<(unsigned)(N_W / 2048), 256, 0, stream>>>(Wu, wub);
  conv_bf16<<<(unsigned)(N_W / 2048), 256, 0, stream>>>(Wd, wdb);

  dim3 g1(N_EXP * TILES_R, I_DIM / TILE_C);   // 512 x 44
  moe_gateup<<<g1, 256, 0, stream>>>(xb, wgb, wub, counts, offsets, perm, hbuf);

  dim3 g2(N_EXP * TILES_R, H_DIM / TILE_C);   // 512 x 16
  moe_down<<<g2, 256, 0, stream>>>(wdb, counts, offsets, perm, hbuf, out);
}
